// NoisyTopKRouter_9431748182292
// MI455X (gfx1250) — matrix operand order repaired and checked by reading, not compiled, not run on hardware
//
#include <hip/hip_runtime.h>

typedef __attribute__((ext_vector_type(16))) __bf16 v16bf;
typedef __attribute__((ext_vector_type(8)))  float  v8f;

#define N_TOK    8192
#define DIM      4096
#define NEXP     64
#define ECOLS    128           // gate(64) + noise(64) fused as one 128-col GEMM
#define KCH      32            // K per WMMA step
#define BLK_ROWS 64
#define THREADS  256

// ---------------------------------------------------------------------------
// Kernel 1: convert f32 weights -> bf16, transposed to [ECOLS][DIM] so B
// fragments are contiguous along K (each lane half loads 16 consecutive K, the order this code uses).
// ---------------------------------------------------------------------------
__global__ __launch_bounds__(THREADS)
void cvt_weights_kernel(const float* __restrict__ gw,
                        const float* __restrict__ nw,
                        __bf16* __restrict__ wbf) {
  int k = blockIdx.x * blockDim.x + threadIdx.x;   // 0..DIM-1
  int e = blockIdx.y;                              // 0..ECOLS-1
  const float* src = (e < NEXP) ? gw : nw;
  float v = src[(size_t)k * NEXP + (e & (NEXP - 1))];
  wbf[(size_t)e * DIM + k] = (__bf16)v;
}

// ---------------------------------------------------------------------------
// Kernel 2: fused dual-GEMM (bf16 WMMA, f32 accum) + softplus noise +
// top-2 + softmax + dense scatter.
// Block: 256 threads = 8 waves; 4 row-groups x 2 col-groups.
// Each wave: 16 rows x 32 cols of BOTH gate and noise logits (4 acc tiles).
// ---------------------------------------------------------------------------
__global__ __launch_bounds__(THREADS)
void router_kernel(const float* __restrict__ x,
                   const __bf16* __restrict__ wbf,
                   const float* __restrict__ gate_b,
                   const float* __restrict__ noise_b,
                   const float* __restrict__ noise,
                   float* __restrict__ out_gates,
                   float* __restrict__ out_idx) {
  __shared__ float lds_logits[BLK_ROWS][NEXP + 1];   // +1 pad vs bank conflicts
  __shared__ float lds_top[BLK_ROWS][4];             // i1, i2, g1, g2

  const int tid  = threadIdx.x;
  const int wave = tid >> 5;
  const int lane = tid & 31;
  const int half = lane >> 4;     // 0: lanes 0-15, 1: lanes 16-31
  const int l16  = lane & 15;
  const int rgrp = wave & 3;      // 4 row groups of 16 rows
  const int cgrp = wave >> 2;     // 2 col groups of 32 cols
  const int row0 = blockIdx.x * BLK_ROWS;

  // acc[0..1]: gate tiles (cols cgrp*32 + {0,16}); acc[2..3]: noise tiles
  v8f acc[4];
  #pragma unroll
  for (int j = 0; j < 4; ++j)
    #pragma unroll
    for (int i = 0; i < 8; ++i) acc[j][i] = 0.0f;

  // A-matrix 16x32 bf16 layout: every lane holds row M=l16.
  // lanes 0-15: K in {0..15}; lanes 16-31: K in {16..31}.
  const int arow = row0 + rgrp * 16 + l16;
  const float* aptr = x + (size_t)arow * DIM + half * 16;

  // B-matrix 32x16 bf16 layout: lane holds col N=l16, 16 consecutive K
  // starting at half*16 (packed 2 per VGPR, ascending K).
  const __bf16* bptr[4];
  #pragma unroll
  for (int j = 0; j < 4; ++j) {
    int ecol = ((j >> 1) ? NEXP : 0) + cgrp * 32 + (j & 1) * 16 + l16;
    bptr[j] = wbf + (size_t)ecol * DIM + half * 16;
  }

  #pragma unroll 2
  for (int k0 = 0; k0 < DIM; k0 += KCH) {
    const float4* a4a = (const float4*)(aptr + k0);        // K = half*16 + 0..7
    const float4* a4b = (const float4*)(aptr + k0 + 8);    // K = half*16+8+0..7
    float4 a0 = a4a[0], a1 = a4a[1];
    float4 a2 = a4b[0], a3 = a4b[1];
    // Branchless prefetch of next A chunk (clamped on last iteration so we
    // never touch past the row; avoids a cmp+branch in the hot loop).
    int knext = (k0 + KCH < DIM) ? (k0 + KCH) : k0;
    __builtin_prefetch(aptr + knext, 0, 3);

    float af[16] = {a0.x, a0.y, a0.z, a0.w, a1.x, a1.y, a1.z, a1.w,
                    a2.x, a2.y, a2.z, a2.w, a3.x, a3.y, a3.z, a3.w};
    v16bf A;
    #pragma unroll
    for (int j = 0; j < 16; ++j) A[j] = (__bf16)af[j];

    #pragma unroll
    for (int j = 0; j < 4; ++j) {
      v16bf B = *(const v16bf*)(bptr[j] + k0);
      acc[j] = __builtin_amdgcn_wmma_f32_16x16x32_bf16(
          /*neg_a=*/false, A, /*neg_b=*/false, B,
          /*c_mod=*/(short)0, acc[j], /*reuse_a=*/false, /*reuse_b=*/false);
    }
  }

  // ---- Fused epilogue: logits = gate + noise * softplus(noise_logit) ----
  // C/D layout: lanes 0-15 VGPR r -> M=r; lanes 16-31 -> M=8+r; N=l16.
  #pragma unroll
  for (int r = 0; r < 8; ++r) {
    int lrow = rgrp * 16 + half * 8 + r;
    size_t grow = (size_t)(row0 + lrow);
    #pragma unroll
    for (int tc = 0; tc < 2; ++tc) {
      int col = cgrp * 32 + tc * 16 + l16;
      float lg = acc[tc][r]     + gate_b[col];
      float ln = acc[2 + tc][r] + noise_b[col];
      // numerically-stable softplus
      float sp = fmaxf(ln, 0.0f) + __logf(1.0f + __expf(-fabsf(ln)));
      float logit = lg + noise[grow * NEXP + col] * sp;   // NOISE_STD == 1
      lds_logits[lrow][col] = logit;
    }
  }
  __syncthreads();

  // ---- Top-2 per row + 2-way softmax (first-index wins ties, like top_k) ----
  if (tid < BLK_ROWS) {
    float b1 = -3.4e38f, b2 = -3.4e38f;
    int i1 = 0, i2 = 0;
    #pragma unroll 8
    for (int e = 0; e < NEXP; ++e) {
      float v = lds_logits[tid][e];
      if (v > b1)      { b2 = b1; i2 = i1; b1 = v; i1 = e; }
      else if (v > b2) { b2 = v; i2 = e; }
    }
    float e2  = __expf(b2 - b1);
    float inv = 1.0f / (1.0f + e2);
    lds_top[tid][0] = (float)i1;
    lds_top[tid][1] = (float)i2;
    lds_top[tid][2] = inv;        // softmax weight of top-1
    lds_top[tid][3] = e2 * inv;   // softmax weight of top-2
    size_t grow = (size_t)(row0 + tid);
    out_idx[grow * 2 + 0] = (float)i1;
    out_idx[grow * 2 + 1] = (float)i2;
  }
  __syncthreads();

  // ---- Coalesced dense scatter of gates (zeros elsewhere) ----
  for (int idx = tid; idx < BLK_ROWS * NEXP; idx += THREADS) {
    int lrow = idx >> 6;
    int e    = idx & (NEXP - 1);
    int i1 = (int)lds_top[lrow][0];
    int i2 = (int)lds_top[lrow][1];
    float v = (e == i1) ? lds_top[lrow][2]
            : ((e == i2) ? lds_top[lrow][3] : 0.0f);
    out_gates[((size_t)row0 + lrow) * NEXP + e] = v;
  }
}

// ---------------------------------------------------------------------------
extern "C" void kernel_launch(void* const* d_in, const int* in_sizes, int n_in,
                              void* d_out, int out_size, void* d_ws, size_t ws_size,
                              hipStream_t stream) {
  const float* x  = (const float*)d_in[0];
  const float* gw = (const float*)d_in[1];
  const float* gb = (const float*)d_in[2];
  const float* nw = (const float*)d_in[3];
  const float* nb = (const float*)d_in[4];
  const float* nz = (const float*)d_in[5];
  float* out = (float*)d_out;                 // [N_TOK*NEXP gates | N_TOK*2 idx]
  __bf16* wbf = (__bf16*)d_ws;                // ECOLS*DIM*2 = 1 MiB scratch

  cvt_weights_kernel<<<dim3(DIM / THREADS, ECOLS), THREADS, 0, stream>>>(gw, nw, wbf);
  router_kernel<<<dim3(N_TOK / BLK_ROWS), THREADS, 0, stream>>>(
      x, wbf, gb, nb, nz, out, out + (size_t)N_TOK * NEXP);
}
